// IPEXLowbitGemmAddInt4_47407849013389
// MI455X (gfx1250) — compile-verified, matrix operand order checked by reading, not run on hardware
//
#include <hip/hip_runtime.h>
#include <hip/hip_bf16.h>

// ---- vector types ----
typedef __attribute__((ext_vector_type(16))) _Float16 v16h;
typedef __attribute__((ext_vector_type(8)))  _Float16 v8h;
typedef __attribute__((ext_vector_type(8)))  float    v8f;
typedef __attribute__((ext_vector_type(4)))  float    v4f;
typedef __attribute__((ext_vector_type(4)))  unsigned int v4u;
typedef __attribute__((ext_vector_type(4)))  int      v4i;

// ---- tiling ----
#define MT 128          // M tile per block
#define NT 128          // N tile per block
#define KT 64           // K step per iteration (two 16x16x32 WMMA k-chunks)
#define ASTR 72         // LDS row stride (halfs): 144B -> 16B aligned, bank-conflict free
#define BSTR 72         // LDS col stride (halfs)
#define GROUP 128       // GPTQ group size

struct V4U2 { v4u a, b; };
union FragU { V4U2 u; v16h h; };

// One 16x32 f16 WMMA operand fragment from LDS (two contiguous 8-half runs
// per the CDNA5 16-bit operand layout).
__device__ __forceinline__ v16h frag16(const _Float16* p, int o0, int o1) {
    FragU f;
    f.u.a = *(const v4u*)(p + o0);
    f.u.b = *(const v4u*)(p + o1);
    return f.h;
}

// ---- register staging ----
struct ARegs { v4f f[8]; };                  // 32 fp32 activations
struct BRegs { v4i q; v4f s; int zw; };      // 4 qweight words + scales + zero word

__device__ __forceinline__ ARegs ldA(const float* gp) {
    ARegs r;
#pragma unroll
    for (int i = 0; i < 8; ++i) r.f[i] = ((const v4f*)gp)[i];
    return r;
}

__device__ __forceinline__ void stA(const ARegs& r, _Float16* sA) {
#pragma unroll
    for (int i = 0; i < 4; ++i) {
        v4f f0 = r.f[2 * i], f1 = r.f[2 * i + 1];
        v8h h;
        h[0] = (_Float16)f0[0]; h[1] = (_Float16)f0[1];
        h[2] = (_Float16)f0[2]; h[3] = (_Float16)f0[3];
        h[4] = (_Float16)f1[0]; h[5] = (_Float16)f1[1];
        h[6] = (_Float16)f1[2]; h[7] = (_Float16)f1[3];
        *(v8h*)(sA + 8 * i) = h;
    }
}

__device__ __forceinline__ BRegs ldB(const int* __restrict__ qweight,
                                     const float* __restrict__ scales,
                                     const int* __restrict__ qzeros,
                                     int k0, int bR, int gcol,
                                     long long N, long long Nq8) {
    BRegs r;
    const int g = k0 >> 7;   // group index (KT=64 never crosses a 128 group)
    r.q  = *(const v4i*)(qweight + (long long)((k0 >> 3) + bR) * N + gcol);
    r.s  = *(const v4f*)(scales + (long long)g * N + gcol);
    r.zw = qzeros[(long long)g * Nq8 + (gcol >> 3)];
    return r;
}

__device__ __forceinline__ void stB(const BRegs& r, _Float16* sB, int bCw, int bR) {
    const int zsh = bCw & 4;   // nibble base within zword (bCw % 8 is 0 or 4)
#pragma unroll
    for (int j = 0; j < 4; ++j) {
        const int   qw = r.q[j];
        const float z  = (float)((r.zw >> (4 * (zsh + j))) & 0xF);
        const float s  = r.s[j];
        v8h h;
#pragma unroll
        for (int i = 0; i < 8; ++i) {
            const float q = (float)((qw >> (4 * i)) & 0xF);
            h[i] = (_Float16)((q - z) * s);
        }
        *(v8h*)(sB + (bCw + j) * BSTR + bR * 8) = h;
    }
}

// 2x4 sub-tiles of 16x16, over one KT=64 LDS tile (two 32-deep WMMA chunks)
__device__ __forceinline__ void computeTile(const _Float16* bufA, const _Float16* bufB,
                                            v8f acc[2][4],
                                            int waveM, int waveN, int laneM, int hiH) {
#pragma unroll
    for (int kk = 0; kk < KT; kk += 32) {
        const int o0 = kk + hiH * 8;
        const int o1 = kk + 16 + hiH * 8;

        v16h af[2];
#pragma unroll
        for (int mi = 0; mi < 2; ++mi)
            af[mi] = frag16(bufA + (waveM * 32 + mi * 16 + laneM) * ASTR, o0, o1);

        v16h bf[4];
#pragma unroll
        for (int ni = 0; ni < 4; ++ni)
            bf[ni] = frag16(bufB + (waveN * 64 + ni * 16 + laneM) * BSTR, o0, o1);

#pragma unroll
        for (int mi = 0; mi < 2; ++mi)
#pragma unroll
            for (int ni = 0; ni < 4; ++ni)
                acc[mi][ni] = __builtin_amdgcn_wmma_f32_16x16x32_f16(
                    false, af[mi], false, bf[ni],
                    (short)0, acc[mi][ni], false, false);
    }
}

__global__ __launch_bounds__(256)
void gptq_int4_wmma_kernel(const float* __restrict__ A,        // [M, K] fp32
                           const float* __restrict__ residual, // [M, N] fp32
                           const int*   __restrict__ qweight,  // [K/8, N] nibbles along K
                           const float* __restrict__ scales,   // [K/GROUP, N]
                           const int*   __restrict__ qzeros,   // [K/GROUP, N/8] nibbles along N
                           const float* __restrict__ bias,     // [N]
                           float* __restrict__ out,            // [M, N] fp32
                           int M, int N, int K) {
    // Double-buffered tiles: 2 x (18 KB + 18 KB) = 72 KB (CDNA5 WGP has 320 KB)
    __shared__ _Float16 ldsA[2][MT * ASTR];
    __shared__ _Float16 ldsB[2][NT * BSTR];

    const int tid   = threadIdx.x;
    const int lane  = tid & 31;
    const int wave  = tid >> 5;
    const int laneM = lane & 15;
    const int hiH   = lane >> 4;
    const int waveM = wave & 3;        // 4 waves along M -> 32 rows each
    const int waveN = wave >> 2;       // 2 waves along N -> 64 cols each

    const int n0 = blockIdx.x * NT;
    const int m0 = blockIdx.y * MT;

    // global staging maps
    const int aRow = tid >> 1;
    const int aKh  = (tid & 1) * 32;
    const int bR   = tid >> 5;          // qweight word-row within K step (0..7)
    const int bCw  = (tid & 31) * 4;    // 4 columns within N tile
    const int gcol = n0 + bCw;

    const long long Nq8 = (long long)(N >> 3);
    const float* gAp = A + (long long)(m0 + aRow) * K + aKh;

    v8f acc[2][4];
#pragma unroll
    for (int mi = 0; mi < 2; ++mi)
#pragma unroll
        for (int ni = 0; ni < 4; ++ni)
            acc[mi][ni] = (v8f){};

    // ---- prologue: stage K-step 0 into buffer 0 ----
    {
        ARegs ar = ldA(gAp);
        BRegs br = ldB(qweight, scales, qzeros, 0, bR, gcol, N, Nq8);
        stA(ar, &ldsA[0][aRow * ASTR + aKh]);
        stB(br, ldsB[0], bCw, bR);
    }
    __syncthreads();

    // ---- pipelined main loop: one barrier per K-step ----
    int cur = 0;
    for (int k0 = KT; k0 < K; k0 += KT) {
        ARegs ar = ldA(gAp + k0);                                    // loads in flight
        BRegs br = ldB(qweight, scales, qzeros, k0, bR, gcol, N, Nq8);

        computeTile(ldsA[cur], ldsB[cur], acc, waveM, waveN, laneM, hiH);  // WMMA on cur

        stA(ar, &ldsA[cur ^ 1][aRow * ASTR + aKh]);                  // VALU co-executes
        stB(br, ldsB[cur ^ 1], bCw, bR);
        __syncthreads();
        cur ^= 1;
    }
    computeTile(ldsA[cur], ldsB[cur], acc, waveM, waveN, laneM, hiH);

    // ---- epilogue: + bias + residual, store ----
    const int mBase = m0 + waveM * 32;
    const int nBase = n0 + waveN * 64;
#pragma unroll
    for (int mi = 0; mi < 2; ++mi) {
#pragma unroll
        for (int ni = 0; ni < 4; ++ni) {
            const v8f c = acc[mi][ni];
            const int col = nBase + ni * 16 + laneM;
            const float bv = bias[col];
            const long long row0 = (long long)(mBase + mi * 16 + hiH * 8);
            const float* resp = residual + row0 * N + col;
            float*       outp = out      + row0 * N + col;
#pragma unroll
            for (int r = 0; r < 8; ++r) {
                outp[(long long)r * N] = c[r] + bv + resp[(long long)r * N];
            }
        }
    }
}

extern "C" void kernel_launch(void* const* d_in, const int* in_sizes, int n_in,
                              void* d_out, int out_size, void* d_ws, size_t ws_size,
                              hipStream_t stream) {
    const float* inp = (const float*)d_in[0];   // [B,S,K] fp32
    const float* res = (const float*)d_in[1];   // [B,S,N] fp32
    const int*   qw  = (const int*)  d_in[2];   // [K/8, N] int32
    const float* sc  = (const float*)d_in[3];   // [K/GROUP, N] fp32
    const int*   qz  = (const int*)  d_in[4];   // [K/GROUP, N/8] int32
    const float* bs  = (const float*)d_in[5];   // [N] fp32
    float*       out = (float*)d_out;

    const int       N = in_sizes[5];
    const long long K = ((long long)in_sizes[2] * 8) / N;
    const long long M = (long long)in_sizes[0] / K;   // B*S

    dim3 grid(N / NT, (unsigned)(M / MT));
    dim3 block(256);
    gptq_int4_wmma_kernel<<<grid, block, 0, stream>>>(
        inp, res, qw, sc, qz, bs, out, (int)M, N, (int)K);
}
